// GNNEncoder_71571335020554
// MI455X (gfx1250) — matrix operand order repaired and checked by reading, not compiled, not run on hardware
//
#include <hip/hip_runtime.h>

typedef __attribute__((ext_vector_type(16))) __bf16 v16bf;
typedef __attribute__((ext_vector_type(8)))  __bf16 v8bf;
typedef __attribute__((ext_vector_type(4)))  __bf16 v4bf;
typedef __attribute__((ext_vector_type(8)))  float  v8f;

#define DDIM 128
#define WROW 136   // padded LDS row stride (bf16 elems): 272B = 68 dwords -> conflict-free

// ---------------- degree / norm ----------------
__global__ void k_deg_init(float* __restrict__ deg, int n) {
    int i = blockIdx.x * blockDim.x + threadIdx.x;
    if (i < n) deg[i] = 1.0f;                       // self-loop contributes 1
}

__global__ void k_deg_edges(const int* __restrict__ dst, float* __restrict__ deg, int e) {
    int i = blockIdx.x * blockDim.x + threadIdx.x;
    if (i < e) atomicAdd(&deg[dst[i]], 1.0f);
}

__global__ void k_dinv(float* __restrict__ deg, int n) {
    int i = blockIdx.x * blockDim.x + threadIdx.x;
    if (i < n) deg[i] = rsqrtf(deg[i]);             // deg >= 1 always
}

// ---------------- fp32 -> bf16 (+optional ReLU) ----------------
__global__ void k_cvt_h(const float* __restrict__ src, __bf16* __restrict__ dst,
                        int n4, int do_relu) {
    int i = blockIdx.x * blockDim.x + threadIdx.x;
    if (i >= n4) return;
    float4 v = ((const float4*)src)[i];
    if (do_relu) {
        v.x = fmaxf(v.x, 0.f); v.y = fmaxf(v.y, 0.f);
        v.z = fmaxf(v.z, 0.f); v.w = fmaxf(v.w, 0.f);
    }
    v4bf o;
    o[0] = (__bf16)v.x; o[1] = (__bf16)v.y; o[2] = (__bf16)v.z; o[3] = (__bf16)v.w;
    ((v4bf*)dst)[i] = o;
}

// ---------------- W (KxN row-major) -> Wt (NxK row-major, bf16) ----------------
__global__ void k_cvt_w(const float* __restrict__ W, __bf16* __restrict__ Wt) {
    int i = blockIdx.x * blockDim.x + threadIdx.x;   // D*D threads
    if (i >= DDIM * DDIM) return;
    int k = i >> 7, n = i & 127;
    Wt[n * DDIM + k] = (__bf16)W[k * DDIM + n];
}

// ---------------- GEMM: C[N x 128] = A[N x 128](bf16) @ W, fp32 acc ----------------
// Wt staged once per workgroup in LDS (padded rows); one wave = 16x128 slab,
// 8 f32 accumulators, K=128 as 4 chunks of 32 via v_wmma_f32_16x16x32_bf16.
__global__ void __launch_bounds__(256)
k_gemm(const __bf16* __restrict__ A, const __bf16* __restrict__ Wt,
       float* __restrict__ C, int ntiles) {
    __shared__ __bf16 lds[DDIM * WROW];
    int tid = threadIdx.x;

    // stage 128x128 bf16 (32KB) into LDS: 2048 chunks of 8 elems, 8 per thread
#pragma unroll
    for (int j = 0; j < 8; j++) {
        int c = tid + j * 256;
        int row = c >> 4, col = (c & 15) * 8;
        *(v8bf*)(lds + row * WROW + col) = *(const v8bf*)(Wt + row * DDIM + col);
    }
    __syncthreads();

    int tile = blockIdx.x * 8 + (tid >> 5);
    if (tile >= ntiles) return;                      // wave-uniform: EXEC stays all-1s
    int lane = tid & 31;
    int l15  = lane & 15;
    int half = lane >> 4;                            // 0 or 1

    const __bf16* arow = A + (size_t)(tile * 16 + l15) * DDIM;

    v8f acc[8];
#pragma unroll
    for (int n = 0; n < 8; n++)
#pragma unroll
        for (int j = 0; j < 8; j++) acc[n][j] = 0.0f;

#pragma unroll
    for (int kc = 0; kc < 4; kc++) {
        // A 16x32 bf16: lane holds row (lane&15); elems 0..7 = K kc*32+half*8+j,
        // elems 8..15 = K kc*32+16+half*8+j  -> two contiguous 16B global loads.
        union { v16bf v; v8bf h[2]; } a;
        a.h[0] = *(const v8bf*)(arow + kc * 32 + half * 8);
        a.h[1] = *(const v8bf*)(arow + kc * 32 + 16 + half * 8);
#pragma unroll
        for (int n = 0; n < 8; n++) {
            // B 32x16 bf16: lane holds column (lane&15), same K order as A (from LDS)
            union { v16bf v; v8bf h[2]; } bu;
            bu.h[0] = *(const v8bf*)(lds + (n * 16 + l15) * WROW + kc * 32 + half * 8);
            bu.h[1] = *(const v8bf*)(lds + (n * 16 + l15) * WROW + kc * 32 + 16 + half * 8);
            v16bf b = bu.v;
            acc[n] = __builtin_amdgcn_wmma_f32_16x16x32_bf16(
                false, a.v, false, b, (short)0, acc[n], false, false);
        }
    }

    // C/D layout: VGPR v -> row half*8+v, col = n*16 + (lane&15)
    int rbase = tile * 16 + half * 8;
#pragma unroll
    for (int n = 0; n < 8; n++) {
#pragma unroll
        for (int v = 0; v < 8; v++) {
            C[(size_t)(rbase + v) * DDIM + n * 16 + l15] = acc[n][v];
        }
    }
}

// ---------------- agg = dinv[i]^2 * hW[i] + bias  (self-loop + bias, full overwrite) ----------------
__global__ void __launch_bounds__(256)
k_agg_init(const float* __restrict__ hW, const float* __restrict__ dinv,
           const float* __restrict__ bias, float* __restrict__ agg, int n32) {
    int t = blockIdx.x * blockDim.x + threadIdx.x;   // N*32 threads, float4 each
    if (t >= n32) return;
    int node = __builtin_amdgcn_readfirstlane(t >> 5);   // wave-uniform -> scalar load
    int c = t & 31;
    float s = dinv[node]; s *= s;
    float4 v = ((const float4*)hW)[t];
    float4 b = ((const float4*)bias)[c];
    float4 o;
    o.x = v.x * s + b.x; o.y = v.y * s + b.y;
    o.z = v.z * s + b.z; o.w = v.w * s + b.w;
    ((float4*)agg)[t] = o;
}

// ---------------- edge scatter: agg[dst] += dinv[src]*dinv[dst] * hW[src] ----------------
// One wave per edge: indices scalarized (s_load), 32 lanes x float4 = 128 features,
// coalesced gather + global_atomic_add_f32; hW/agg are L2-resident (25.6MB << 192MB).
__global__ void __launch_bounds__(256)
k_scatter(const int* __restrict__ srcI, const int* __restrict__ dstI,
          const float* __restrict__ dinv, const float* __restrict__ hW,
          float* __restrict__ agg, int nth) {
    int t = blockIdx.x * blockDim.x + threadIdx.x;   // E*32 threads
    if (t >= nth) return;
    int e = __builtin_amdgcn_readfirstlane(t >> 5);  // wave-uniform edge id
    int c = t & 31;
    int s = srcI[e], d = dstI[e];                    // scalar loads
    float nrm = dinv[s] * dinv[d];                   // scalar
    float4 v = ((const float4*)hW)[s * 32 + c];      // saddr + lane offset
    float* ap = agg + (size_t)d * DDIM + c * 4;
    atomicAdd(ap + 0, v.x * nrm);
    atomicAdd(ap + 1, v.y * nrm);
    atomicAdd(ap + 2, v.z * nrm);
    atomicAdd(ap + 3, v.w * nrm);
}

extern "C" void kernel_launch(void* const* d_in, const int* in_sizes, int n_in,
                              void* d_out, int out_size, void* d_ws, size_t ws_size,
                              hipStream_t stream) {
    const float* x  = (const float*)d_in[0];
    const int*   ei = (const int*)d_in[1];
    const float* W[3] = { (const float*)d_in[2], (const float*)d_in[4], (const float*)d_in[6] };
    const float* B[3] = { (const float*)d_in[3], (const float*)d_in[5], (const float*)d_in[7] };

    const int N = in_sizes[0] / DDIM;
    const int E = in_sizes[1] / 2;
    const int* srcI = ei;
    const int* dstI = ei + E;
    const int ntiles = N / 16;                       // N = 50000 -> 3125 exact

    // workspace carve-up (256B aligned)
    char* p = (char*)d_ws;
    auto carve = [&](size_t bytes) { char* r = p; p += (bytes + 255) & ~(size_t)255; return r; };
    float*  deg  = (float*)carve((size_t)N * 4);              // becomes dinv in place
    float*  hW   = (float*)carve((size_t)N * DDIM * 4);
    float*  aggb = (float*)carve((size_t)N * DDIM * 4);
    __bf16* hbf  = (__bf16*)carve((size_t)N * DDIM * 2);
    __bf16* Wt   = (__bf16*)carve((size_t)DDIM * DDIM * 2);
    (void)ws_size; (void)n_in; (void)out_size;

    const int T = 256;
    // normalization coefficients
    k_deg_init <<<(N + T - 1) / T, T, 0, stream>>>(deg, N);
    k_deg_edges<<<(E + T - 1) / T, T, 0, stream>>>(dstI, deg, E);
    k_dinv     <<<(N + T - 1) / T, T, 0, stream>>>(deg, N);

    const float* hsrc = x;
    const int n4  = N * DDIM / 4;
    const int n32 = N * 32;
    const int eth = E * 32;
    for (int l = 0; l < 3; l++) {
        k_cvt_h<<<(n4 + T - 1) / T, T, 0, stream>>>(hsrc, hbf, n4, l > 0 ? 1 : 0);
        k_cvt_w<<<(DDIM * DDIM + T - 1) / T, T, 0, stream>>>(W[l], Wt);
        k_gemm <<<(ntiles + 7) / 8, T, 0, stream>>>(hbf, Wt, hW, ntiles);
        float* agg = (l == 2) ? (float*)d_out : aggb;
        k_agg_init<<<(n32 + T - 1) / T, T, 0, stream>>>(hW, deg, B[l], agg, n32);
        k_scatter <<<(eth + T - 1) / T, T, 0, stream>>>(srcI, dstI, deg, hW, agg, eth);
        hsrc = agg;
    }
}
